// PatchPositionEmbedding_56487409877802
// MI455X (gfx1250) — compile-verified, matrix operand order checked by reading, not run on hardware
//
#include <hip/hip_runtime.h>
#include <hip/hip_bf16.h>

typedef __attribute__((ext_vector_type(8)))  float   v8f;
typedef __attribute__((ext_vector_type(8)))  __bf16  v8bf;
typedef __attribute__((ext_vector_type(16))) __bf16  v16bf;

#define BM 128
#define BN 128
#define BK 32
#define KPAD 40          // 32 data + 8 pad shorts -> 80B row stride (16B aligned, bank staggered)
#define KSTEPS 24        // 768 / 32
#define NPATCH 196
#define SEQ 197
#define DMODEL 768
#define CIMG (3*224*224) // floats per image
#define CPLANE (224*224)

union Frag { v16bf v; v8bf h[2]; };

// Split f0,f1 into bf16 hi/lo pairs, packed as {f1_part[31:16], f0_part[31:16]}.
// All 32-bit integer math + v_perm_b32 packing (no 16-bit register ops).
static __device__ __forceinline__ void cvt_pair(float f0, float f1,
                                                unsigned& hp, unsigned& lp) {
    unsigned u0 = __float_as_uint(f0);
    unsigned u1 = __float_as_uint(f1);
    unsigned uh0 = u0 + 0x7FFFu + ((u0 >> 16) & 1u);   // RNE-rounded bf16 in [31:16]
    unsigned uh1 = u1 + 0x7FFFu + ((u1 >> 16) & 1u);
    hp = __builtin_amdgcn_perm(uh1, uh0, 0x07060302u); // {uh1.hi16, uh0.hi16}
    float r0 = f0 - __uint_as_float(uh0 & 0xFFFF0000u);
    float r1 = f1 - __uint_as_float(uh1 & 0xFFFF0000u);
    unsigned v0 = __float_as_uint(r0);
    unsigned v1 = __float_as_uint(r1);
    unsigned vl0 = v0 + 0x7FFFu + ((v0 >> 16) & 1u);
    unsigned vl1 = v1 + 0x7FFFu + ((v1 >> 16) & 1u);
    lp = __builtin_amdgcn_perm(vl1, vl0, 0x07060302u);
}

__global__ __launch_bounds__(256)
void vit_patch_embed_gemm(const float* __restrict__ x,
                          const float* __restrict__ Wt,
                          const float* __restrict__ bias,
                          const float* __restrict__ pos,
                          float* __restrict__ out)
{
    __shared__ __attribute__((aligned(16))) unsigned short Ah[2][BM][KPAD];
    __shared__ __attribute__((aligned(16))) unsigned short Al[2][BM][KPAD];
    __shared__ __attribute__((aligned(16))) unsigned short Bh[2][BN][KPAD];
    __shared__ __attribute__((aligned(16))) unsigned short Bl[2][BN][KPAD];

    const int tid  = threadIdx.x;
    const int lane = tid & 31;
    const int wid  = tid >> 5;       // 0..7
    const int wm   = wid >> 1;       // 0..3 : 32 M-rows per wave
    const int wn   = wid & 1;        // 0..1 : 64 N-cols per wave

    const int tileM = blockIdx.x;    // 0..391
    const int tileN = blockIdx.y;    // 0..5

    // ---- staging mapping: one 64B chunk (16 floats) per thread ----
    const int lrow = tid >> 1;       // 0..127
    const int hch  = tid & 1;        // which 16-float half of the 32-K slab

    // im2col address precompute for this thread's A row
    const int am  = tileM * BM + lrow;          // global patch row (b*196 + p)
    const int ab  = am / NPATCH;
    const int ap  = am - ab * NPATCH;
    const int agi = ap / 14;
    const int agj = ap - agi * 14;
    const int abase = ab * CIMG + agi * (16 * 224) + agj * 16;  // float index into x

    const int bnrow = tileN * BN + lrow;        // global output channel (B row)
    const int wbase = bnrow * DMODEL;

    const int koff = (lane & 16) ? 8 : 0;       // ISA bf16 A/B fragment K grouping
    const int rsel = lane & 15;

    float4 ra[4], rb[4];

    auto loadA = [&](int ks) {
        const int k0 = ks * BK;
        const int c  = k0 >> 8;                       // channel
        const int ph = ((k0 & 255) >> 4) + hch;       // row within patch
        const float4* p = reinterpret_cast<const float4*>(x + abase + c * CPLANE + ph * 224);
        ra[0] = p[0]; ra[1] = p[1]; ra[2] = p[2]; ra[3] = p[3];
    };
    auto loadB = [&](int ks) {
        const float4* p = reinterpret_cast<const float4*>(Wt + wbase + ks * BK + hch * 16);
        rb[0] = p[0]; rb[1] = p[1]; rb[2] = p[2]; rb[3] = p[3];
    };

    auto stage = [&](int buf) {
        unsigned hA[8], lA[8], hB[8], lB[8];
        #pragma unroll
        for (int q = 0; q < 4; ++q) {
            cvt_pair(ra[q].x, ra[q].y, hA[2*q],     lA[2*q]);
            cvt_pair(ra[q].z, ra[q].w, hA[2*q + 1], lA[2*q + 1]);
            cvt_pair(rb[q].x, rb[q].y, hB[2*q],     lB[2*q]);
            cvt_pair(rb[q].z, rb[q].w, hB[2*q + 1], lB[2*q + 1]);
        }
        uint4* p;
        p = reinterpret_cast<uint4*>(&Ah[buf][lrow][hch * 16]);
        p[0] = make_uint4(hA[0], hA[1], hA[2], hA[3]);
        p[1] = make_uint4(hA[4], hA[5], hA[6], hA[7]);
        p = reinterpret_cast<uint4*>(&Al[buf][lrow][hch * 16]);
        p[0] = make_uint4(lA[0], lA[1], lA[2], lA[3]);
        p[1] = make_uint4(lA[4], lA[5], lA[6], lA[7]);
        p = reinterpret_cast<uint4*>(&Bh[buf][lrow][hch * 16]);
        p[0] = make_uint4(hB[0], hB[1], hB[2], hB[3]);
        p[1] = make_uint4(hB[4], hB[5], hB[6], hB[7]);
        p = reinterpret_cast<uint4*>(&Bl[buf][lrow][hch * 16]);
        p[0] = make_uint4(lB[0], lB[1], lB[2], lB[3]);
        p[1] = make_uint4(lB[4], lB[5], lB[6], lB[7]);
    };

    auto ldfrag = [&](const unsigned short* rowp) -> v16bf {
        Frag f;
        f.h[0] = *reinterpret_cast<const v8bf*>(rowp + koff);        // K koff..koff+7  -> VGPR 0-3
        f.h[1] = *reinterpret_cast<const v8bf*>(rowp + koff + 16);   // K koff+16..+23 -> VGPR 4-7
        return f.v;
    };

    v8f acc[2][4] = {};

    auto domm = [&](int buf) {
        v16bf afh[2], afl[2], bfh[4], bfl[4];
        #pragma unroll
        for (int i = 0; i < 2; ++i) {
            const int r = wm * 32 + i * 16 + rsel;
            afh[i] = ldfrag(&Ah[buf][r][0]);
            afl[i] = ldfrag(&Al[buf][r][0]);
        }
        #pragma unroll
        for (int j = 0; j < 4; ++j) {
            const int c = wn * 64 + j * 16 + rsel;
            bfh[j] = ldfrag(&Bh[buf][c][0]);
            bfl[j] = ldfrag(&Bl[buf][c][0]);
        }
        #pragma unroll
        for (int i = 0; i < 2; ++i)
            #pragma unroll
            for (int j = 0; j < 4; ++j) {
                acc[i][j] = __builtin_amdgcn_wmma_f32_16x16x32_bf16(
                    false, afh[i], false, bfh[j], (short)0, acc[i][j], false, false);
                acc[i][j] = __builtin_amdgcn_wmma_f32_16x16x32_bf16(
                    false, afl[i], false, bfh[j], (short)0, acc[i][j], false, false);
                acc[i][j] = __builtin_amdgcn_wmma_f32_16x16x32_bf16(
                    false, afh[i], false, bfl[j], (short)0, acc[i][j], false, false);
            }
    };

    // ---- prologue ----
    loadA(0); loadB(0);
    stage(0);                 // buf0 <- k-step 0
    loadA(1); loadB(1);       // regs <- k-step 1
    __syncthreads();

    // ---- main loop: one barrier per K-step; ds-stores overlap WMMA ----
    for (int ks = 0; ks < KSTEPS; ++ks) {
        const int buf = ks & 1;
        if (ks + 1 < KSTEPS) stage(buf ^ 1);     // buffer last read at ks-1 (safe after prior sync)
        if (ks + 2 < KSTEPS) { loadA(ks + 2); loadB(ks + 2); }
        domm(buf);
        __syncthreads();
    }

    // ---- epilogue: + bias + pos_emb, scatter into [B, 197, 768] ----
    #pragma unroll
    for (int j = 0; j < 4; ++j) {
        const int gn = tileN * BN + wn * 64 + j * 16 + rsel;
        const float bv = bias[gn];
        #pragma unroll
        for (int i = 0; i < 2; ++i) {
            const int gm0 = tileM * BM + wm * 32 + i * 16 + ((lane & 16) ? 8 : 0);
            #pragma unroll
            for (int v = 0; v < 8; ++v) {
                const int gm  = gm0 + v;
                const int obb = gm / NPATCH;
                const int opp = gm - obb * NPATCH;
                out[(obb * SEQ + 1 + opp) * DMODEL + gn] =
                    acc[i][j][v] + bv + pos[(1 + opp) * DMODEL + gn];
            }
        }
    }
}

__global__ __launch_bounds__(256)
void vit_cls_rows(const float* __restrict__ ct, const float* __restrict__ pos,
                  float* __restrict__ out)
{
    const int t = blockIdx.x * blockDim.x + threadIdx.x;
    if (t >= 256 * DMODEL) return;
    const int b = t / DMODEL;
    const int n = t - b * DMODEL;
    out[(b * SEQ) * DMODEL + n] = ct[n] + pos[n];
}

extern "C" void kernel_launch(void* const* d_in, const int* in_sizes, int n_in,
                              void* d_out, int out_size, void* d_ws, size_t ws_size,
                              hipStream_t stream) {
    const float* x   = (const float*)d_in[0];   // [256,3,224,224]
    const float* W   = (const float*)d_in[1];   // [768,768]
    const float* b   = (const float*)d_in[2];   // [768]
    const float* ct  = (const float*)d_in[3];   // [768]
    const float* pos = (const float*)d_in[4];   // [197,768]
    float* out = (float*)d_out;                 // [256,197,768]

    dim3 grid(392, 6);                          // 50176/128 x 768/128
    vit_patch_embed_gemm<<<grid, dim3(256), 0, stream>>>(x, W, b, pos, out);
    vit_cls_rows<<<dim3((256 * DMODEL + 255) / 256), dim3(256), 0, stream>>>(ct, pos, out);
}
